// GatConv_48524540510778
// MI455X (gfx1250) — matrix operand order repaired and checked by reading, not compiled, not run on hardware
//
#include <hip/hip_runtime.h>

#define NN   8192
#define INC  256
#define HIDC 256

typedef __bf16 v16bf __attribute__((ext_vector_type(16)));
typedef __bf16 v8bf  __attribute__((ext_vector_type(8)));
typedef __bf16 v4bf  __attribute__((ext_vector_type(4)));
typedef float  v8f   __attribute__((ext_vector_type(8)));
typedef float  v4f   __attribute__((ext_vector_type(4)));

// ---------------------------------------------------------------------------
// k0: wT[c][k] = bf16(weight[k][c])   (256x256, makes B-fragments contiguous)
// ---------------------------------------------------------------------------
__global__ __launch_bounds__(256) void k0_transpose_w(const float* __restrict__ w,
                                                      __bf16* __restrict__ wT) {
    int k = blockIdx.x;   // 0..255
    int c = threadIdx.x;  // 0..255
    wT[(size_t)c * INC + k] = (__bf16)w[(size_t)k * INC + c];
}

// ---------------------------------------------------------------------------
// k1: h = nodes @ weight via WMMA bf16; emit hT (bf16, [HID][N]) and f1,f2.
// Block = 128 thr (4 waves), 16 rows x 256 cols; wave w -> 64 cols.
// K-loop software-pipelined with a 2-slot register double buffer so WMMAs
// never wait on same-iteration loads.
// ---------------------------------------------------------------------------
__global__ __launch_bounds__(128) void k1_gemm(const float* __restrict__ nodes,
                                               const __bf16* __restrict__ wT,
                                               const float* __restrict__ cw1,
                                               const float* __restrict__ cw2,
                                               __bf16* __restrict__ hT,
                                               float* __restrict__ f1,
                                               float* __restrict__ f2) {
    __shared__ float red[4][2][16];

    const int tid  = threadIdx.x;
    const int wv   = tid >> 5;
    const int lane = tid & 31;
    const int n    = lane & 15;   // A-row / B-col / C-col within tile
    const int hs   = lane >> 4;   // lane-half selects K / M ranges
    const int rb   = blockIdx.x;  // 16-row block
    const int row  = rb * 16 + n;
    const int cb   = wv * 64;

    v8f acc[4];
#pragma unroll
    for (int t = 0; t < 4; ++t)
#pragma unroll
        for (int i = 0; i < 8; ++i) acc[t][i] = 0.0f;

    // 2-slot pipeline registers
    v4f  a0[2], a1[2], a2[2], a3[2];
    v8bf wb0[2][4], wb1[2][4];

    auto loadslot = [&](int s, int k) {
        const v4f* ap = (const v4f*)(nodes + (size_t)row * INC + k + hs * 8);
        a0[s] = ap[0]; a1[s] = ap[1];   // K chunk 1 (8 floats)
        a2[s] = ap[4]; a3[s] = ap[5];   // K chunk 2 at +16 floats
#pragma unroll
        for (int t = 0; t < 4; ++t) {
            const v8bf* bp = (const v8bf*)(wT + (size_t)(cb + t * 16 + n) * INC + k + hs * 8);
            wb0[s][t] = bp[0];
            wb1[s][t] = bp[2];
        }
    };

    auto kstep = [&](int cur, int nxt, int kn) {
        loadslot(nxt, kn);  // issue next k-step's loads first
        v16bf A;
#pragma unroll
        for (int i = 0; i < 4; ++i) {
            A[i]      = (__bf16)a0[cur][i];
            A[i + 4]  = (__bf16)a1[cur][i];
            A[i + 8]  = (__bf16)a2[cur][i];
            A[i + 12] = (__bf16)a3[cur][i];
        }
#pragma unroll
        for (int t = 0; t < 4; ++t) {
            v16bf B;
#pragma unroll
            for (int i = 0; i < 8; ++i) { B[i] = wb0[cur][t][i]; B[i + 8] = wb1[cur][t][i]; }
            acc[t] = __builtin_amdgcn_wmma_f32_16x16x32_bf16(
                false, A, false, B, (short)0, acc[t], false, false);
        }
    };

    loadslot(0, 0);
    for (int k = 0; k < INC; k += 64) {
        kstep(0, 1, k + 32);
        kstep(1, 0, (k + 64 < INC) ? (k + 64) : 0);
    }

    // ---- write hT (bf16) + partial f1/f2 from f32 accumulators ----
    float w1c[4], w2c[4];
#pragma unroll
    for (int t = 0; t < 4; ++t) {
        w1c[t] = cw1[cb + t * 16 + n];
        w2c[t] = cw2[cb + t * 16 + n];
    }
    float pf1[8], pf2[8];
#pragma unroll
    for (int kk = 0; kk < 8; ++kk) { pf1[kk] = 0.0f; pf2[kk] = 0.0f; }

#pragma unroll
    for (int t = 0; t < 4; ++t) {
        v8bf hv;
#pragma unroll
        for (int kk = 0; kk < 8; ++kk) {
            float v = acc[t][kk];          // h[rb*16 + hs*8 + kk][cb + t*16 + n]
            hv[kk] = (__bf16)v;
            pf1[kk] += v * w1c[t];
            pf2[kk] += v * w2c[t];
        }
        *(v8bf*)(hT + (size_t)(cb + t * 16 + n) * NN + rb * 16 + hs * 8) = hv;
    }

    // reduce across the 16 lanes of each half (xor bits 0..3 stay in-half)
#pragma unroll
    for (int m = 1; m <= 8; m <<= 1)
#pragma unroll
        for (int kk = 0; kk < 8; ++kk) {
            pf1[kk] += __shfl_xor(pf1[kk], m, 32);
            pf2[kk] += __shfl_xor(pf2[kk], m, 32);
        }
    if (n == 0) {  // lanes 0 and 16 of each wave
#pragma unroll
        for (int kk = 0; kk < 8; ++kk) {
            red[wv][0][kk + hs * 8] = pf1[kk];
            red[wv][1][kk + hs * 8] = pf2[kk];
        }
    }
    __syncthreads();
    if (tid < 16) {
        float s1 = 0.0f, s2 = 0.0f;
#pragma unroll
        for (int w = 0; w < 4; ++w) { s1 += red[w][0][tid]; s2 += red[w][1][tid]; }
        f1[rb * 16 + tid] = s1;
        f2[rb * 16 + tid] = s2;
    }
}

// ---------------------------------------------------------------------------
// k2: fused attention: p = exp(lrelu(f1_i + f2_j) + adj[i][j]); out = (p@h)/rowsum + bias
// Block = 128 thr (4 waves), 16 rows; j-loop in steps of 32, unrolled x2 with a
// 2-slot register double buffer for adj/f2/B so every WMMA consumes data loaded
// one tile earlier (no same-iteration load waits). LDS p-tile double-buffered
// on the same slot index; one barrier per tile; global_prefetch_b8 feeds HBM.
// ---------------------------------------------------------------------------
__global__ __launch_bounds__(128) void k2_attn(const float* __restrict__ adj,
                                               const __bf16* __restrict__ hT,
                                               const float* __restrict__ f1,
                                               const float* __restrict__ f2,
                                               const float* __restrict__ bias,
                                               float* __restrict__ out) {
    __shared__ __align__(16) __bf16 plds[2][16][32];
    __shared__ float slds[128];
    __shared__ float rsum[16];

    const int tid  = threadIdx.x;
    const int wv   = tid >> 5;
    const int lane = tid & 31;
    const int n    = lane & 15;
    const int hs   = lane >> 4;
    const int rb   = blockIdx.x;
    const int cb   = wv * 64;

    // p-producer mapping: thread -> (row, 4 cols), coalesced 128B per row
    const int prow = tid >> 3;         // 0..15
    const int pcol = (tid & 7) << 2;   // 0,4,...,28
    const float f1r = f1[rb * 16 + prow];
    const float* arow = adj + (size_t)(rb * 16 + prow) * NN + pcol;
    const float* f2p  = f2 + pcol;

    v8f acc[4];
#pragma unroll
    for (int t = 0; t < 4; ++t)
#pragma unroll
        for (int i = 0; i < 8; ++i) acc[t][i] = 0.0f;

    float sump = 0.0f;

    // 2-slot pipeline registers: adj tile, f2 tile, all 4 B fragments
    v4f  av[2], fv[2];
    v8bf b0[2][4], b1[2][4];

    auto loadslot = [&](int s, int j) {
        av[s] = *(const v4f*)(arow + j);
        fv[s] = *(const v4f*)(f2p + j);
#pragma unroll
        for (int t = 0; t < 4; ++t) {
            const v8bf* bp = (const v8bf*)(hT + (size_t)(cb + t * 16 + n) * NN + j + hs * 8);
            b0[s][t] = bp[0];
            b1[s][t] = bp[2];
        }
    };

    auto step = [&](int j, int cur, int nxt, int jn) {
        loadslot(nxt, jn);                          // issue next tile's loads first
        __builtin_prefetch(arow + j + 512, 0, 1);   // global_prefetch_b8, 16 tiles ahead

        // produce p for current tile (data loaded one tile ago)
        v4bf pv;
#pragma unroll
        for (int e = 0; e < 4; ++e) {
            float s = f1r + fv[cur][e];
            s = (s > 0.0f) ? s : 0.2f * s;   // leaky_relu
            s += av[cur][e];                  // 0 or -1e9 (mask) -> exp underflows to 0
            float p = __expf(s);
            sump += p;
            pv[e] = (__bf16)p;
        }
        *(v4bf*)&plds[cur][prow][pcol] = pv;
        __syncthreads();  // includes s_wait_dscnt; slot-indexed buffer => 1 barrier/tile

        // A fragment (p-tile): row n, K = [8hs,8hs+8) u [16+8hs, ...)
        const v8bf* apod = (const v8bf*)&plds[cur][n][hs * 8];
        v8bf pa0 = apod[0];
        v8bf pa1 = apod[2];  // +16 bf16
        v16bf A;
#pragma unroll
        for (int i = 0; i < 8; ++i) { A[i] = pa0[i]; A[i + 8] = pa1[i]; }

#pragma unroll
        for (int t = 0; t < 4; ++t) {
            v16bf B;
#pragma unroll
            for (int i = 0; i < 8; ++i) { B[i] = b0[cur][t][i]; B[i + 8] = b1[cur][t][i]; }
            acc[t] = __builtin_amdgcn_wmma_f32_16x16x32_bf16(
                false, A, false, B, (short)0, acc[t], false, false);
        }
    };

    loadslot(0, 0);
    for (int jb = 0; jb < NN; jb += 64) {
        step(jb,      0, 1, jb + 32);
        step(jb + 32, 1, 0, (jb + 64 < NN) ? (jb + 64) : 0);
    }

    // softmax denominator per row
    slds[tid] = sump;
    __syncthreads();
    if (tid < 16) {
        float s = 0.0f;
#pragma unroll
        for (int e = 0; e < 8; ++e) s += slds[tid * 8 + e];
        rsum[tid] = s;
    }
    __syncthreads();

    float rinv[8];
#pragma unroll
    for (int kk = 0; kk < 8; ++kk) rinv[kk] = 1.0f / rsum[kk + hs * 8];

#pragma unroll
    for (int t = 0; t < 4; ++t) {
        int c = cb + t * 16 + n;
        float bv = bias[c];
#pragma unroll
        for (int kk = 0; kk < 8; ++kk) {
            out[(size_t)(rb * 16 + kk + hs * 8) * HIDC + c] = acc[t][kk] * rinv[kk] + bv;
        }
    }
}

// ---------------------------------------------------------------------------
extern "C" void kernel_launch(void* const* d_in, const int* in_sizes, int n_in,
                              void* d_out, int out_size, void* d_ws, size_t ws_size,
                              hipStream_t stream) {
    const float* nodes  = (const float*)d_in[0];
    const float* adj    = (const float*)d_in[1];
    const float* weight = (const float*)d_in[2];
    const float* cw1    = (const float*)d_in[3];
    const float* cw2    = (const float*)d_in[4];
    const float* bias   = (const float*)d_in[5];
    float* out = (float*)d_out;

    char* ws = (char*)d_ws;
    __bf16* wT = (__bf16*)(ws);                               // 256*256*2   = 128 KB
    __bf16* hT = (__bf16*)(ws + (128u << 10));                // 256*8192*2  = 4 MB
    float*  f1 = (float*)(ws + (128u << 10) + (4u << 20));    // 32 KB
    float*  f2 = (float*)(ws + (128u << 10) + (4u << 20) + (32u << 10));

    k0_transpose_w<<<256, 256, 0, stream>>>(weight, wT);
    k1_gemm<<<NN / 16, 128, 0, stream>>>(nodes, wT, cw1, cw2, hT, f1, f2);
    k2_attn<<<NN / 16, 128, 0, stream>>>(adj, hT, f1, f2, bias, out);
}
